// OrthoPQ_15315853378150
// MI455X (gfx1250) — compile-verified, matrix operand order checked by reading, not run on hardware
//
#include <hip/hip_runtime.h>

// ---------------------------------------------------------------------------
// OrthoPQ forward for MI455X (gfx1250, wave32, WMMA).
// Heavy math on v_wmma_f32_16x16x32_bf16; workgroup-cooperative LDS staging
// of the shared B tiles (8x reduction of L2 traffic vs per-wave loads).
// ---------------------------------------------------------------------------

typedef __bf16 bf16_t;
typedef bf16_t v16bf __attribute__((ext_vector_type(16)));
typedef bf16_t v8bf  __attribute__((ext_vector_type(8)));
typedef float  v8f   __attribute__((ext_vector_type(8)));

#define N_ROWS   1024
#define IN_F     2048
#define OUT_F    4096
#define BOOKS    8
#define LW       256     // len_word == words == 256
#define SC_F     30.0f
#define MARGIN_F 0.5f

// LDS B-tile row pitch in ushorts: 256 + 8 pad so fragment reads hit banks at
// stride 4 (row*528B/4 = 132*row -> bank 4*row mod 64) instead of all bank 0.
#define BPITCH   264

// ---- helpers ---------------------------------------------------------------

__device__ __forceinline__ unsigned short f2bf(float f) {
  // round-to-nearest-even f32 -> bf16
  unsigned int u = __float_as_uint(f);
  u += 0x7FFFu + ((u >> 16) & 1u);
  return (unsigned short)(u >> 16);
}

__device__ __forceinline__ unsigned int pack2(float x, float y) {
  return (unsigned int)f2bf(x) | ((unsigned int)f2bf(y) << 16);
}

__device__ __forceinline__ float wave_sum(float v) {
#pragma unroll
  for (int m = 16; m > 0; m >>= 1) v += __shfl_xor(v, m, 32);
  return v;
}

__device__ __forceinline__ float wave_max(float v) {
#pragma unroll
  for (int m = 16; m > 0; m >>= 1) v = fmaxf(v, __shfl_xor(v, m, 32));
  return v;
}

// 16x32 bf16 fragment from row-major [rows, 256] global memory.
// ISA 16-bit A layout: lane L (0..15) = row M=L, K {0..7, 16..23};
// lane L+16 = row M=L, K {8..15, 24..31}.  B (stored N x K row-major) loads
// with the identical pattern.
__device__ __forceinline__ v16bf load_frag(const unsigned short* __restrict__ base,
                                           int ld, int row0, int k0, int lane) {
  const int r = lane & 15;
  const int h = lane >> 4;
  const unsigned short* p = base + (size_t)(row0 + r) * ld + k0 + h * 8;
  v16bf f;
  ((v8bf*)&f)[0] = *(const v8bf*)(p);
  ((v8bf*)&f)[1] = *(const v8bf*)(p + 16);
  return f;
}

// Same fragment pattern out of the padded LDS B tile.
__device__ __forceinline__ v16bf load_frag_lds(const unsigned short* base,
                                               int row0, int k0, int lane) {
  const int r = lane & 15;
  const int h = lane >> 4;
  const unsigned short* p = base + (row0 + r) * BPITCH + k0 + h * 8;
  v16bf f;
  ((v8bf*)&f)[0] = *(const v8bf*)(p);
  ((v8bf*)&f)[1] = *(const v8bf*)(p + 16);
  return f;
}

// ---- stage 1: row L2-normalize f32 [rows,256] -> bf16 ----------------------

__global__ void norm_rows_kernel(const float* __restrict__ in,
                                 unsigned short* __restrict__ out, int rows) {
  const int wid  = blockIdx.x * (blockDim.x >> 5) + (threadIdx.x >> 5);
  if (wid >= rows) return;                  // wave-uniform
  const int lane = threadIdx.x & 31;
  const float* rp = in + (size_t)wid * LW + lane * 8;
  float4 a = ((const float4*)rp)[0];
  float4 b = ((const float4*)rp)[1];
  float s = a.x*a.x + a.y*a.y + a.z*a.z + a.w*a.w +
            b.x*b.x + b.y*b.y + b.z*b.z + b.w*b.w;
  s = wave_sum(s);
  const float inv = 1.0f / fmaxf(sqrtf(s), 1e-12f);
  uint4 pk;
  pk.x = pack2(a.x*inv, a.y*inv); pk.y = pack2(a.z*inv, a.w*inv);
  pk.z = pack2(b.x*inv, b.y*inv); pk.w = pack2(b.z*inv, b.w*inv);
  *(uint4*)(out + (size_t)wid * LW + lane * 8) = pk;
}

// ---- stage 2: split input into per-book chunks; raw + normalized bf16 ------

__global__ void split_x_kernel(const float* __restrict__ in,
                               unsigned short* __restrict__ xraw,
                               unsigned short* __restrict__ xn) {
  const int wid  = blockIdx.x * 8 + (threadIdx.x >> 5);  // b*1024 + n
  const int lane = threadIdx.x & 31;
  const int b = wid >> 10, n = wid & 1023;
  const float* rp = in + (size_t)n * IN_F + b * LW + lane * 8;
  float4 a = ((const float4*)rp)[0];
  float4 c = ((const float4*)rp)[1];
  float s = a.x*a.x + a.y*a.y + a.z*a.z + a.w*a.w +
            c.x*c.x + c.y*c.y + c.z*c.z + c.w*c.w;
  s = wave_sum(s);
  const float inv = 1.0f / fmaxf(sqrtf(s), 1e-12f);
  const size_t o = (size_t)wid * LW + lane * 8;
  uint4 raw;
  raw.x = pack2(a.x, a.y); raw.y = pack2(a.z, a.w);
  raw.z = pack2(c.x, c.y); raw.w = pack2(c.z, c.w);
  *(uint4*)(xraw + o) = raw;
  uint4 nn;
  nn.x = pack2(a.x*inv, a.y*inv); nn.y = pack2(a.z*inv, a.w*inv);
  nn.z = pack2(c.x*inv, c.y*inv); nn.w = pack2(c.z*inv, c.w*inv);
  *(uint4*)(xn + o) = nn;
}

// ---- stage 3: transpose mlp -> [b][w][k] bf16, convert codebooks -> bf16 ---

__global__ void prep_small_kernel(const float* __restrict__ mlp,
                                  const float* __restrict__ cb,
                                  unsigned short* __restrict__ mlpT,
                                  unsigned short* __restrict__ cbb) {
  const int TOT = BOOKS * LW * LW;
  const int tid = blockIdx.x * blockDim.x + threadIdx.x;
  if (tid < TOT) {
    const int b = tid >> 16;
    const int w = (tid >> 8) & 255;
    const int k = tid & 255;
    mlpT[tid] = f2bf(mlp[(size_t)b * (LW * LW) + (size_t)k * LW + w]);
  } else if (tid < 2 * TOT) {
    const int i = tid - TOT;
    cbb[i] = f2bf(cb[i]);   // codebooks[b][d][w] is already N(=d) x K(=w)
  }
}

// ---- stage 4/6: per-book GEMM  C[b] = A[b](1024x256) * B[b](256x256)^T -----
// Block = 8 waves; block tile M=128 x N=64; B slab (64x256 bf16) staged in LDS
// once and shared by all 8 waves.

__global__ void gemm_book_kernel(const unsigned short* __restrict__ A,
                                 const unsigned short* __restrict__ B,
                                 float* __restrict__ C) {
  __shared__ unsigned short Bt[64 * BPITCH];          // 33 KB
  const int t  = threadIdx.x;
  const int bi = blockIdx.x;        // 0..255
  const int b  = bi >> 5;           // 32 blocks per book
  const int mb = (bi >> 2) & 7;     // 8 M-blocks of 128
  const int n0 = (bi & 3) * 64;     // 4 N-tiles of 64
  const unsigned short* Ab = A + (size_t)b * N_ROWS * LW;
  const unsigned short* Bb = B + (size_t)b * LW * LW;

  // cooperative B slab load: 64 rows x 256 ushorts (2048 uint4, 8 per thread)
  for (int i = t; i < 64 * 32; i += 256) {
    const int row = i >> 5, ch = i & 31;
    *(uint4*)(&Bt[row * BPITCH + ch * 8]) =
        *(const uint4*)(Bb + (size_t)(n0 + row) * LW + ch * 8);
  }
  __syncthreads();

  const int lane = t & 31;
  const int m0 = mb * 128 + (t >> 5) * 16;
  v8f acc[4] = {};
  for (int k0 = 0; k0 < LW; k0 += 32) {
    const v16bf afrag = load_frag(Ab, LW, m0, k0, lane);
#pragma unroll
    for (int j = 0; j < 4; ++j) {
      const v16bf bfrag = load_frag_lds(Bt, j * 16, k0, lane);
      acc[j] = __builtin_amdgcn_wmma_f32_16x16x32_bf16(
          false, afrag, false, bfrag, (short)0, acc[j], false, false);
    }
  }
  const int colb  = n0 + (lane & 15);
  const int rbase = m0 + (lane >> 4) * 8;
  float* Cb = C + (size_t)b * N_ROWS * LW;
#pragma unroll
  for (int j = 0; j < 4; ++j)
#pragma unroll
    for (int r = 0; r < 8; ++r)
      Cb[(size_t)(rbase + r) * LW + colb + j * 16] = acc[j][r];
}

// ---- stage 5: softmax over 256 words per (b,n) row -------------------------

__global__ void softmax_kernel(const float* __restrict__ logits,
                               unsigned short* __restrict__ xcb,
                               float* __restrict__ out3) {
  const int wid  = blockIdx.x * 8 + (threadIdx.x >> 5);  // b*1024 + n
  const int lane = threadIdx.x & 31;
  const float* rp = logits + (size_t)wid * LW + lane * 8;
  float4 a = ((const float4*)rp)[0];
  float4 c = ((const float4*)rp)[1];
  float v[8] = {a.x, a.y, a.z, a.w, c.x, c.y, c.z, c.w};
  float mx = v[0];
#pragma unroll
  for (int i = 1; i < 8; ++i) mx = fmaxf(mx, v[i]);
  mx = wave_max(mx);
  float s = 0.f;
#pragma unroll
  for (int i = 0; i < 8; ++i) { v[i] = __expf(v[i] - mx); s += v[i]; }
  s = wave_sum(s);
  const float inv = 1.0f / s;
#pragma unroll
  for (int i = 0; i < 8; ++i) v[i] *= inv;
  const size_t o = (size_t)wid * LW + lane * 8;
  uint4 pk;
  pk.x = pack2(v[0], v[1]); pk.y = pack2(v[2], v[3]);
  pk.z = pack2(v[4], v[5]); pk.w = pack2(v[6], v[7]);
  *(uint4*)(xcb + o) = pk;
  const int b = wid >> 10, n = wid & 1023;
  float* op = out3 + (size_t)n * (BOOKS * LW) + b * LW + lane * 8;
  ((float4*)op)[0] = make_float4(v[0], v[1], v[2], v[3]);
  ((float4*)op)[1] = make_float4(v[4], v[5], v[6], v[7]);
}

// ---- stage 8: fused cos GEMMs (xn,sn) x wn^T + clip/scale/margin -----------
// Block = 8 waves; block tile M=128 x N=64 for BOTH streams; the wn slab is
// staged in LDS once and reused 16x (8 waves x 2 streams).

__global__ void cos_gemm_kernel(const unsigned short* __restrict__ xn,
                                const unsigned short* __restrict__ sn,
                                const unsigned short* __restrict__ wn,
                                const long long* __restrict__ label,
                                float* __restrict__ out) {
  __shared__ unsigned short Bt[64 * BPITCH];          // 33 KB
  const int t  = threadIdx.x;
  const int bi = blockIdx.x;        // 0..4095
  const int b  = bi >> 9;           // 512 blocks per book
  const int mb = (bi >> 6) & 7;     // 8 M-blocks of 128
  const int n0 = (bi & 63) * 64;    // 64 N-tiles of 64
  const unsigned short* Ax = xn + (size_t)b * N_ROWS * LW;
  const unsigned short* As = sn + (size_t)b * N_ROWS * LW;
  const unsigned short* Bb = wn + (size_t)b * OUT_F * LW;

  for (int i = t; i < 64 * 32; i += 256) {
    const int row = i >> 5, ch = i & 31;
    *(uint4*)(&Bt[row * BPITCH + ch * 8]) =
        *(const uint4*)(Bb + (size_t)(n0 + row) * LW + ch * 8);
  }
  __syncthreads();

  const int lane = t & 31;
  const int m0 = mb * 128 + (t >> 5) * 16;
  v8f accX[4] = {};
  v8f accS[4] = {};
  for (int k0 = 0; k0 < LW; k0 += 32) {
    const v16bf a1 = load_frag(Ax, LW, m0, k0, lane);
    const v16bf a2 = load_frag(As, LW, m0, k0, lane);
#pragma unroll
    for (int j = 0; j < 4; ++j) {
      const v16bf bfrag = load_frag_lds(Bt, j * 16, k0, lane);
      accX[j] = __builtin_amdgcn_wmma_f32_16x16x32_bf16(
          false, a1, false, bfrag, (short)0, accX[j], false, false);
      accS[j] = __builtin_amdgcn_wmma_f32_16x16x32_bf16(
          false, a2, false, bfrag, (short)0, accS[j], false, false);
    }
  }
  const int colb  = n0 + (lane & 15);
  const int rbase = m0 + (lane >> 4) * 8;
  float* o1 = out;
  float* o2 = out + (size_t)N_ROWS * BOOKS * OUT_F;
#pragma unroll
  for (int r = 0; r < 8; ++r) {
    const int row = rbase + r;
    const long long lab = label[row];
    const size_t base = (size_t)row * (BOOKS * OUT_F) + (size_t)b * OUT_F;
#pragma unroll
    for (int j = 0; j < 4; ++j) {
      const int col = colb + j * 16;
      const float m  = (lab == (long long)col) ? (SC_F * MARGIN_F) : 0.f;
      const float vx = fminf(fmaxf(accX[j][r], -1.f), 1.f) * SC_F - m;
      const float vs = fminf(fmaxf(accS[j][r], -1.f), 1.f) * SC_F - m;
      o1[base + col] = vx;
      o2[base + col] = vs;
    }
  }
}

// ---------------------------------------------------------------------------

extern "C" void kernel_launch(void* const* d_in, const int* in_sizes, int n_in,
                              void* d_out, int out_size, void* d_ws, size_t ws_size,
                              hipStream_t stream) {
  (void)in_sizes; (void)n_in; (void)out_size; (void)ws_size;
  const float*     input  = (const float*)d_in[0];
  const long long* label  = (const long long*)d_in[1];
  const float*     weight = (const float*)d_in[2];
  const float*     mlp    = (const float*)d_in[3];
  const float*     cb     = (const float*)d_in[4];
  float* out = (float*)d_out;

  char* ws = (char*)d_ws;
  size_t off = 0;
  auto carve = [&](size_t bytes) -> char* {
    char* p = ws + off;
    off += (bytes + 255) & ~(size_t)255;
    return p;
  };
  unsigned short* wn   = (unsigned short*)carve((size_t)BOOKS * OUT_F * LW * 2);  // 16 MB
  unsigned short* xraw = (unsigned short*)carve((size_t)BOOKS * N_ROWS * LW * 2); //  4 MB
  unsigned short* xnb  = (unsigned short*)carve((size_t)BOOKS * N_ROWS * LW * 2); //  4 MB
  unsigned short* mlpT = (unsigned short*)carve((size_t)BOOKS * LW * LW * 2);     //  1 MB
  unsigned short* cbb  = (unsigned short*)carve((size_t)BOOKS * LW * LW * 2);     //  1 MB
  float*          logit= (float*)carve((size_t)BOOKS * N_ROWS * LW * 4);          //  8 MB
  unsigned short* xcb  = (unsigned short*)carve((size_t)BOOKS * N_ROWS * LW * 2); //  4 MB
  float*          sbuf = (float*)carve((size_t)BOOKS * N_ROWS * LW * 4);          //  8 MB
  unsigned short* snb  = (unsigned short*)carve((size_t)BOOKS * N_ROWS * LW * 2); //  4 MB

  // 1) wn = l2norm(weight) -> bf16        (32768 rows, 1 wave/row)
  norm_rows_kernel<<<(BOOKS * OUT_F) / 8, 256, 0, stream>>>(weight, wn, BOOKS * OUT_F);
  // 2) x chunks: raw + normalized bf16    (8192 rows)
  split_x_kernel<<<(BOOKS * N_ROWS) / 8, 256, 0, stream>>>(input, xraw, xnb);
  // 3) mlp transpose + codebooks convert
  prep_small_kernel<<<(2 * BOOKS * LW * LW + 255) / 256, 256, 0, stream>>>(mlp, cb, mlpT, cbb);
  // 4) logits = x @ mlp                   (WMMA, LDS-staged B)
  gemm_book_kernel<<<256, 256, 0, stream>>>(xraw, mlpT, logit);
  // 5) xc_soft = softmax(logits); write f32 output + bf16 for next GEMM
  softmax_kernel<<<(BOOKS * N_ROWS) / 8, 256, 0, stream>>>(
      logit, xcb, out + 2ull * N_ROWS * BOOKS * OUT_F);
  // 6) s = xc_soft @ codebooks^T          (WMMA, LDS-staged B)
  gemm_book_kernel<<<256, 256, 0, stream>>>(xcb, cbb, sbuf);
  // 7) sn = l2norm(s) -> bf16
  norm_rows_kernel<<<(BOOKS * N_ROWS) / 8, 256, 0, stream>>>(sbuf, snb, BOOKS * N_ROWS);
  // 8) fused cos GEMMs + clip/scale/margin epilogue (WMMA, LDS-staged shared B)
  cos_gemm_kernel<<<4096, 256, 0, stream>>>(xnb, snb, wn, label, out);
}
